// simple_attn_19413252178674
// MI455X (gfx1250) — compile-verified, matrix operand order checked by reading, not run on hardware
//
#include <hip/hip_runtime.h>
#include <hip/hip_bf16.h>
#include <math.h>

typedef __attribute__((ext_vector_type(16))) __bf16 bf16x16;
typedef __attribute__((ext_vector_type(8)))  float  floatx8;

#define CCH   256
#define NPIX  16384   // 128*128
#define HEADS 8
#define DH    32
#define TP    64      // pixels per workgroup tile
#define LDX   272     // row stride (bf16 elems) for [pix][C] tiles (16B-aligned rows)
#define LDKV  528     // row stride for k|v staging [pix][512]
#define LDT   68      // row stride for transposed kT/vT tiles [256][64]
#define LDKVT 48      // row stride for kv^T tile [256][32]
#define LDXF  264     // row stride (f32) for xf residual tile

// ---- WMMA fragment helpers (wave32, ISA 7.12.2 layouts) -------------------

// A/B fragment from a row-major [rows][ld] bf16 tile (LDS or global):
// lane L holds row (row0 + L%16); elems 0..7 = cols col0+8*(L/16)+i,
// elems 8..15 = cols col0+16+8*(L/16)+i.
__device__ __forceinline__ bf16x16 ldfrag(const __bf16* p0, int ld, int row0, int col0) {
  const int lane = threadIdx.x & 31;
  const int hl   = lane >> 4;
  const __bf16* p = p0 + (size_t)(row0 + (lane & 15)) * ld + col0 + 8 * hl;
  bf16x16 f;
#pragma unroll
  for (int i = 0; i < 8; ++i) f[i] = p[i];
#pragma unroll
  for (int i = 0; i < 8; ++i) f[8 + i] = p[16 + i];
  return f;
}

__device__ __forceinline__ floatx8 wmma_bf16(bf16x16 a, bf16x16 b, floatx8 c) {
  // (neg_a, A, neg_b, B, c_mod, C, reuse_a, reuse_b)
  return __builtin_amdgcn_wmma_f32_16x16x32_bf16(false, a, false, b, (short)0, c,
                                                 false, false);
}

// ---- Prep: weights -> bf16, zero kv accumulator ---------------------------

__global__ __launch_bounds__(256) void k_prep(const float* __restrict__ qkvw,
                                              const float* __restrict__ o1w,
                                              const float* __restrict__ o2w,
                                              __bf16* __restrict__ qkvwb,
                                              __bf16* __restrict__ o1wb,
                                              __bf16* __restrict__ o2wb,
                                              float* __restrict__ kvacc) {
  int i = blockIdx.x * 256 + threadIdx.x;
  if (i < 768 * 256) qkvwb[i] = (__bf16)qkvw[i];
  if (i < 256 * 256) {
    o1wb[i] = (__bf16)o1w[i];
    o2wb[i] = (__bf16)o2w[i];
    kvacc[i] = 0.0f;   // 8 batches * 8 heads * 32*32 == 65536 floats
  }
}

// ---- Kernel 1: qkv GEMM + k/v LayerNorm + kv = k^T v / N ------------------

__global__ __launch_bounds__(256) void k_qkv_kv(const float* __restrict__ x,
                                                const __bf16* __restrict__ qkvw,
                                                const float* __restrict__ qkvb,
                                                const float* __restrict__ klnw,
                                                const float* __restrict__ klnb,
                                                const float* __restrict__ vlnw,
                                                const float* __restrict__ vlnb,
                                                __bf16* __restrict__ qout,
                                                float* __restrict__ kvacc) {
  __shared__ __bf16 xs[TP * LDX];        // [pix][C] bf16; reused as kT[256][LDT]
  __shared__ __bf16 kvst[TP * LDKV];     // [pix][ k(0..255) | v(256..511) ]
  __shared__ __bf16 vT[CCH * LDT];       // [h*32+e][pix]

  const int tid  = threadIdx.x;
  const int wv   = tid >> 5;
  const int lane = tid & 31;
  const int hl   = lane >> 4;
  const int nn   = lane & 15;
  const int b    = blockIdx.x >> 8;            // 256 tiles per batch
  const int pix0 = (blockIdx.x & 255) * TP;
  const float* xb = x + (size_t)b * CCH * NPIX;

  // Stage x -> LDS bf16, transposed to [pix][C] (coalesced global reads along pix)
  for (int i = tid; i < TP * CCH; i += 256) {
    int c = i >> 6, p = i & 63;
    xs[p * LDX + c] = (__bf16)xb[(size_t)c * NPIX + pix0 + p];
  }
  __syncthreads();

  // GEMM1: D[pix, oc768] = x_tile @ qkv_w^T   (4 m-tiles x 48 n-tiles, K=256)
  for (int job = wv; job < 192; job += 8) {
    int mt = job & 3, nt = job >> 2;
    floatx8 acc = {};
#pragma unroll
    for (int kt = 0; kt < CCH; kt += 32) {
      bf16x16 a  = ldfrag(xs, LDX, mt * 16, kt);
      bf16x16 bb = ldfrag(qkvw, CCH, nt * 16, kt);   // qkv_w is [oc][k] row-major
      acc = wmma_bf16(a, bb, acc);
    }
    // channel decode: oc = h*96 + kind*32 + d  (head-major q/k/v interleave)
    int h = nt / 6, r6 = nt % 6, kind = r6 >> 1, d0 = (r6 & 1) * 16;
    int oc = nt * 16 + nn;
    float bias = qkvb[oc];
    int cj = h * 32 + d0 + nn;                       // channel in 256-space
    if (kind == 0) {
      __bf16* qrow = qout + (size_t)(b * NPIX + pix0) * CCH;
#pragma unroll
      for (int r = 0; r < 8; ++r) {
        int m = mt * 16 + hl * 8 + r;
        qrow[(size_t)m * CCH + cj] = (__bf16)(acc[r] + bias);
      }
    } else {
      int col = (kind == 1 ? 0 : 256) + cj;
#pragma unroll
      for (int r = 0; r < 8; ++r) {
        int m = mt * 16 + hl * 8 + r;
        kvst[m * LDKV + col] = (__bf16)(acc[r] + bias);
      }
    }
  }
  __syncthreads();

  // LayerNorm over d=32 (ddof=1, /(std+eps)); write transposed per-head tiles
  __bf16* kT = xs;   // x tile is dead; exact size match (64*272 == 256*68)
  for (int g = tid; g < 1024; g += 256) {
    int p = g & 63, h = (g >> 6) & 7, kind = g >> 9;
    const __bf16* src = kvst + p * LDKV + kind * 256 + h * 32;
    float vals[32];
    float s = 0.0f;
#pragma unroll
    for (int j = 0; j < 32; ++j) { vals[j] = (float)src[j]; s += vals[j]; }
    float mean = s * (1.0f / 32.0f);
    float s2 = 0.0f;
#pragma unroll
    for (int j = 0; j < 32; ++j) { float d = vals[j] - mean; s2 += d * d; }
    float inv = 1.0f / (sqrtf(s2 * (1.0f / 31.0f)) + 1e-5f);
    const float* w  = (kind == 0 ? klnw : vlnw) + h * 32;
    const float* bb = (kind == 0 ? klnb : vlnb) + h * 32;
    __bf16* dst = (kind == 0 ? kT : vT);
#pragma unroll
    for (int j = 0; j < 32; ++j) {
      float y = w[j] * ((vals[j] - mean) * inv) + bb[j];
      dst[(h * 32 + j) * LDT + p] = (__bf16)y;
    }
  }
  __syncthreads();

  // kv partial: D[dd,e] = sum_pix kT[dd][pix] * vT[e][pix]; atomic accumulate /N
  for (int job = wv; job < 32; job += 8) {
    int h = job >> 2, mt = (job >> 1) & 1, ntl = job & 1;
    floatx8 acc = {};
#pragma unroll
    for (int kt = 0; kt < TP; kt += 32) {
      bf16x16 a  = ldfrag(kT, LDT, h * 32 + mt * 16, kt);
      bf16x16 bb = ldfrag(vT, LDT, h * 32 + ntl * 16, kt);
      acc = wmma_bf16(a, bb, acc);
    }
    float* dst = kvacc + ((size_t)b * HEADS + h) * (DH * DH);
#pragma unroll
    for (int r = 0; r < 8; ++r) {
      int dd = mt * 16 + hl * 8 + r;
      int e  = ntl * 16 + nn;
      unsafeAtomicAdd(&dst[dd * DH + e], acc[r] * (1.0f / (float)NPIX));
    }
  }
}

// ---- Kernel 2: attn = q@kv, +res, o1+GELU, o2+res -------------------------

__global__ __launch_bounds__(256) void k_attn_mlp(const float* __restrict__ x,
                                                  const __bf16* __restrict__ qbuf,
                                                  const float* __restrict__ kvacc,
                                                  const __bf16* __restrict__ o1w,
                                                  const float* __restrict__ o1b,
                                                  const __bf16* __restrict__ o2w,
                                                  const float* __restrict__ o2b,
                                                  float* __restrict__ out) {
  __shared__ __bf16 kvT[CCH * LDKVT];   // [h*32+e][dd]
  __shared__ __bf16 qs[TP * LDX];       // q tile [pix][C]; reused for h1 (GELU out)
  __shared__ float  xf[TP * LDXF];      // fp32 residual tile [pix][C]
  __shared__ __bf16 rt[TP * LDX];       // ret = attn + xf, bf16 [pix][C]

  const int tid  = threadIdx.x;
  const int wv   = tid >> 5;
  const int lane = tid & 31;
  const int hl   = lane >> 4;
  const int nn   = lane & 15;
  const int b    = blockIdx.x >> 8;
  const int pix0 = (blockIdx.x & 255) * TP;
  const float* xb = x + (size_t)b * CCH * NPIX;

  // kv (fp32 global) -> transposed bf16 LDS tile: kvT[h*32+e][dd]
  for (int i = tid; i < HEADS * DH * DH; i += 256) {
    int h = i >> 10, dd = (i >> 5) & 31, e = i & 31;
    kvT[(h * 32 + e) * LDKVT + dd] =
        (__bf16)kvacc[((size_t)b * HEADS + h) * (DH * DH) + dd * DH + e];
  }
  // q tile
  const __bf16* qrow = qbuf + (size_t)(b * NPIX + pix0) * CCH;
  for (int i = tid; i < TP * CCH; i += 256) {
    int p = i >> 8, c = i & 255;
    qs[p * LDX + c] = qrow[(size_t)p * CCH + c];
  }
  // x tile, fp32, [pix][C]
  for (int i = tid; i < TP * CCH; i += 256) {
    int c = i >> 6, p = i & 63;
    xf[p * LDXF + c] = xb[(size_t)c * NPIX + pix0 + p];
  }
  __syncthreads();

  // attn GEMM: per 16-wide channel tile nt (head = nt/2), K = 32
  for (int job = wv; job < 64; job += 8) {
    int mt = job & 3, nt = job >> 2;
    int h = nt >> 1;
    bf16x16 a  = ldfrag(qs, LDX, mt * 16, h * 32);
    bf16x16 bb = ldfrag(kvT, LDKVT, nt * 16, 0);
    floatx8 acc = {};
    acc = wmma_bf16(a, bb, acc);
    int cc = nt * 16 + nn;
#pragma unroll
    for (int r = 0; r < 8; ++r) {
      int m = mt * 16 + hl * 8 + r;
      rt[m * LDX + cc] = (__bf16)(acc[r] + xf[m * LDXF + cc]);  // residual 1
    }
  }
  __syncthreads();

  // o1 GEMM + exact GELU -> h1 (reuse q tile storage)
  __bf16* h1 = qs;
  for (int job = wv; job < 64; job += 8) {
    int mt = job & 3, nt = job >> 2;
    floatx8 acc = {};
#pragma unroll
    for (int kt = 0; kt < CCH; kt += 32) {
      bf16x16 a  = ldfrag(rt, LDX, mt * 16, kt);
      bf16x16 bb = ldfrag(o1w, CCH, nt * 16, kt);
      acc = wmma_bf16(a, bb, acc);
    }
    int cc = nt * 16 + nn;
    float bias = o1b[cc];
#pragma unroll
    for (int r = 0; r < 8; ++r) {
      int m = mt * 16 + hl * 8 + r;
      float v = acc[r] + bias;
      float g = 0.5f * v * (1.0f + erff(v * 0.70710678118654752f));
      h1[m * LDX + cc] = (__bf16)g;
    }
  }
  __syncthreads();

  // o2 GEMM + bias + residual 2 -> out[b][c][pix]
  float* ob = out + (size_t)b * CCH * NPIX;
  for (int job = wv; job < 64; job += 8) {
    int mt = job & 3, nt = job >> 2;
    floatx8 acc = {};
#pragma unroll
    for (int kt = 0; kt < CCH; kt += 32) {
      bf16x16 a  = ldfrag(h1, LDX, mt * 16, kt);
      bf16x16 bb = ldfrag(o2w, CCH, nt * 16, kt);
      acc = wmma_bf16(a, bb, acc);
    }
    int cc = nt * 16 + nn;
    float bias = o2b[cc];
#pragma unroll
    for (int r = 0; r < 8; ++r) {
      int m = mt * 16 + hl * 8 + r;
      ob[(size_t)cc * NPIX + pix0 + m] = acc[r] + bias + xf[m * LDXF + cc];
    }
  }
}

// ---- Host launcher --------------------------------------------------------

extern "C" void kernel_launch(void* const* d_in, const int* in_sizes, int n_in,
                              void* d_out, int out_size, void* d_ws, size_t ws_size,
                              hipStream_t stream) {
  const float* x     = (const float*)d_in[0];
  const float* qkv_w = (const float*)d_in[1];
  const float* qkv_b = (const float*)d_in[2];
  const float* o1_w  = (const float*)d_in[3];
  const float* o1_b  = (const float*)d_in[4];
  const float* o2_w  = (const float*)d_in[5];
  const float* o2_b  = (const float*)d_in[6];
  const float* kln_w = (const float*)d_in[7];
  const float* kln_b = (const float*)d_in[8];
  const float* vln_w = (const float*)d_in[9];
  const float* vln_b = (const float*)d_in[10];
  float* out = (float*)d_out;

  char* ws = (char*)d_ws;
  __bf16* qkvwb = (__bf16*)(ws);                 // 768*256*2 = 393216 B
  __bf16* o1wb  = (__bf16*)(ws + 393216);        // 131072 B
  __bf16* o2wb  = (__bf16*)(ws + 524288);        // 131072 B
  float*  kvacc = (float*)(ws + 655360);         // 8*8*32*32*4 = 262144 B
  __bf16* qbuf  = (__bf16*)(ws + 917504);        // 8*16384*256*2 = 64 MiB

  k_prep<<<768, 256, 0, stream>>>(qkv_w, o1_w, o2_w, qkvwb, o1wb, o2wb, kvacc);
  k_qkv_kv<<<2048, 256, 0, stream>>>(x, qkvwb, qkv_b, kln_w, kln_b, vln_w, vln_b,
                                     qbuf, kvacc);
  k_attn_mlp<<<2048, 256, 0, stream>>>(x, qbuf, kvacc, o1wb, o1_b, o2wb, o2_b, out);
}
